// AttnPooling_14078902797024
// MI455X (gfx1250) — matrix operand order repaired and checked by reading, not compiled, not run on hardware
//
#include <hip/hip_runtime.h>

typedef __attribute__((ext_vector_type(16))) __bf16 v16bf;
typedef __attribute__((ext_vector_type(8)))  __bf16 v8bf;
typedef __attribute__((ext_vector_type(8)))  float  v8f;

#define IN_FEAT 256
#define DENSE   128
#define NHEADS  8

// ---------------------------------------------------------------------------
// Kernel 0: pack W1 [256,128] and W2 [128,8] (f32, row-major, x@W convention)
// into bf16 WMMA B-operand layout.
//   B 32x16 bf16 tile layout (wave32): lane L -> column n = L%16, half = L/16,
//   element e (0..15) -> K = 8*half + (e&7) + 16*(e>>3).
// W1p: [kt(8)][nt(8)][lane(32)][e(16)]   (K tiles of 32 over IN_FEAT=256)
// W2p: [kt(4)][lane(32)][e(16)]          (N padded 8 -> 16 with zeros)
// ---------------------------------------------------------------------------
__global__ void pack_weights_kernel(const float* __restrict__ W1,
                                    const float* __restrict__ W2,
                                    __bf16* __restrict__ W1p,
                                    __bf16* __restrict__ W2p)
{
    int idx = blockIdx.x * blockDim.x + threadIdx.x;
    if (idx < 8 * 8 * 32 * 16) {
        int e    = idx & 15;
        int lane = (idx >> 4) & 31;
        int nt   = (idx >> 9) & 7;
        int kt   = idx >> 12;
        int k = kt * 32 + (lane >> 4) * 8 + (e & 7) + (e >> 3) * 16;
        int n = nt * 16 + (lane & 15);
        W1p[idx] = (__bf16)W1[k * DENSE + n];
    } else {
        int i = idx - 8 * 8 * 32 * 16;
        if (i < 4 * 32 * 16) {
            int e    = i & 15;
            int lane = (i >> 4) & 31;
            int kt   = i >> 9;
            int k = kt * 32 + (lane >> 4) * 8 + (e & 7) + (e >> 3) * 16;
            int c = lane & 15;
            W2p[i] = (c < NHEADS) ? (__bf16)W2[k * NHEADS + c] : (__bf16)0.0f;
        }
    }
}

// ---------------------------------------------------------------------------
// Kernel 1: scores[node][head] = tanh(h @ W1 + b1) @ W2 + b2
// Block = 256 threads = 8 waves; wave w owns rows [blk*128 + w*16, +16).
//
// Packed W1 (64 KB) is staged block-wide into LDS with CDNA5 async
// load-to-LDS (ASYNCcnt path), then WMMA B operands come from ds_load.
// After GEMM1 the same LDS region is reused (post-barrier) for the bf16
// tanh tiles feeding GEMM2.
//
// A 16x32 bf16 layout: lane L -> row m = L%16, half = L/16;
//   elements 0..7  -> K = k0 + 8*half + e
//   elements 8..15 -> K = k0 + 16 + 8*half + (e-8)
// ---------------------------------------------------------------------------
__global__ __launch_bounds__(256) void score_kernel(
    const float*  __restrict__ h,
    const __bf16* __restrict__ W1p,
    const __bf16* __restrict__ W2p,
    const float*  __restrict__ b1,
    const float*  __restrict__ b2,
    float*        __restrict__ scores,
    int N)
{
    // 64 KB: phase 1 = packed W1 copy; phase 2 = per-wave 16x128 bf16 S tiles
    __shared__ __bf16 w1s[8 * 8 * 32 * 16];

    const int tid  = threadIdx.x;
    const int lane = tid & 31;
    const int wave = tid >> 5;
    const int half = lane >> 4;        // 0 or 1
    const int lrow = lane & 15;        // row within A tile / col within D tile
    const int tileBase = blockIdx.x * 128 + wave * 16;

    // ---- stage packed W1 -> LDS via async copy (16 B per lane per op) ----
    {
        const unsigned lds0 = (unsigned)(uintptr_t)&w1s[0];
#pragma unroll
        for (int it = 0; it < 16; ++it) {
            const int idx = it * 256 + tid;                 // 16-byte chunk id
            const char* src = (const char*)W1p + (size_t)idx * 16;
            const unsigned dst = lds0 + (unsigned)idx * 16;
            asm volatile("global_load_async_to_lds_b128 %0, %1, off"
                         :: "v"(dst), "v"(src) : "memory");
        }
        asm volatile("s_wait_asynccnt 0x0" ::: "memory");
    }
    __syncthreads();

    int row = tileBase + lrow;
    if (row >= N) row = N - 1;         // clamp loads; stores guarded below
    const float* hp = h + (size_t)row * IN_FEAT;

    v8f acc[8] = {};

#pragma unroll
    for (int kt = 0; kt < 8; ++kt) {
        const int c0 = kt * 32 + half * 8;
        const int c1 = kt * 32 + 16 + half * 8;
        float4 f0 = *(const float4*)(hp + c0);
        float4 f1 = *(const float4*)(hp + c0 + 4);
        float4 f2 = *(const float4*)(hp + c1);
        float4 f3 = *(const float4*)(hp + c1 + 4);
        v16bf a;
        a[0]  = (__bf16)f0.x;  a[1]  = (__bf16)f0.y;
        a[2]  = (__bf16)f0.z;  a[3]  = (__bf16)f0.w;
        a[4]  = (__bf16)f1.x;  a[5]  = (__bf16)f1.y;
        a[6]  = (__bf16)f1.z;  a[7]  = (__bf16)f1.w;
        a[8]  = (__bf16)f2.x;  a[9]  = (__bf16)f2.y;
        a[10] = (__bf16)f2.z;  a[11] = (__bf16)f2.w;
        a[12] = (__bf16)f3.x;  a[13] = (__bf16)f3.y;
        a[14] = (__bf16)f3.z;  a[15] = (__bf16)f3.w;
#pragma unroll
        for (int nt = 0; nt < 8; ++nt) {
            v16bf b = *(const v16bf*)(w1s + (size_t)((kt * 8 + nt) * 32 + lane) * 16);
            acc[nt] = __builtin_amdgcn_wmma_f32_16x16x32_bf16(
                false, a, false, b, (short)0, acc[nt], false, false);
        }
    }
    __syncthreads();   // all waves done reading W1 tiles before reuse

    // bias + tanh; stage as bf16 tile S[16][128] per wave (reuses w1s).
    // D layout: lane L, vgpr v -> row m = v + 8*half, col n = nt*16 + lrow.
    __bf16* S = w1s + wave * 16 * DENSE;
#pragma unroll
    for (int nt = 0; nt < 8; ++nt) {
        const int n = nt * 16 + lrow;
        const float bias = b1[n];
#pragma unroll
        for (int v = 0; v < 8; ++v) {
            const int m = v + half * 8;
            S[m * DENSE + n] = (__bf16)tanhf(acc[nt][v] + bias);
        }
    }
    __syncthreads();

    // GEMM2: S(16x128) @ W2p(128x16-padded) -> 16x16, heads in cols 0..7
    v8f acc2 = {};
#pragma unroll
    for (int kt = 0; kt < 4; ++kt) {
        const int c0 = kt * 32 + half * 8;
        const int c1 = kt * 32 + 16 + half * 8;
        v8bf r0 = *(const v8bf*)(S + lrow * DENSE + c0);
        v8bf r1 = *(const v8bf*)(S + lrow * DENSE + c1);
        v16bf a2 = __builtin_shufflevector(r0, r1,
            0, 1, 2, 3, 4, 5, 6, 7, 8, 9, 10, 11, 12, 13, 14, 15);
        v16bf bv = *(const v16bf*)(W2p + (size_t)(kt * 32 + lane) * 16);
        acc2 = __builtin_amdgcn_wmma_f32_16x16x32_bf16(
            false, a2, false, bv, (short)0, acc2, false, false);
    }

    if (lrow < NHEADS) {
        const float bias2 = b2[lrow];
#pragma unroll
        for (int v = 0; v < 8; ++v) {
            const int node = tileBase + v + half * 8;
            if (node < N)
                scores[(size_t)node * NHEADS + lrow] = acc2[v] + bias2;
        }
    }
}

// ---------------------------------------------------------------------------
// Kernel 2: per-graph softmax over nodes (per head), head-average, weighted
// segment sum of h. One block (256 threads) per graph; sorted segment_ids
// give each graph a contiguous node range found by binary search.
// Pass 3 streams h with float4 loads: 64 threads cover one row (256 feats),
// 4 rows in flight per inner iteration.
// ---------------------------------------------------------------------------
__global__ __launch_bounds__(256) void pool_kernel(
    const float* __restrict__ h,
    const int*   __restrict__ seg,
    const float* __restrict__ scores,
    float*       __restrict__ out,
    int N)
{
    const int g    = blockIdx.x;
    const int t    = threadIdx.x;
    const int lane = t & 31;
    const int wave = t >> 5;

    // lower_bound(g), lower_bound(g+1) over sorted segment ids
    int lo = 0, hi = N;
    while (lo < hi) { int mid = (lo + hi) >> 1; if (seg[mid] < g)     lo = mid + 1; else hi = mid; }
    const int start = lo;
    hi = N;
    while (lo < hi) { int mid = (lo + hi) >> 1; if (seg[mid] < g + 1) lo = mid + 1; else hi = mid; }
    const int end = lo;

    __shared__ float red[8][NHEADS];
    __shared__ float smax[NHEADS];
    __shared__ float rden[NHEADS];
    __shared__ float wbuf[256];
    __shared__ float accbuf[256 * 4];

    // ---- pass 1: per-head max ----
    float mx[NHEADS];
#pragma unroll
    for (int hd = 0; hd < NHEADS; ++hd) mx[hd] = -3.402823466e38f;
    for (int i = start + t; i < end; i += 256) {
        const float4* sp = (const float4*)(scores + (size_t)i * NHEADS);
        float4 s0 = sp[0], s1 = sp[1];
        mx[0] = fmaxf(mx[0], s0.x); mx[1] = fmaxf(mx[1], s0.y);
        mx[2] = fmaxf(mx[2], s0.z); mx[3] = fmaxf(mx[3], s0.w);
        mx[4] = fmaxf(mx[4], s1.x); mx[5] = fmaxf(mx[5], s1.y);
        mx[6] = fmaxf(mx[6], s1.z); mx[7] = fmaxf(mx[7], s1.w);
    }
#pragma unroll
    for (int hd = 0; hd < NHEADS; ++hd)
#pragma unroll
        for (int off = 16; off; off >>= 1)
            mx[hd] = fmaxf(mx[hd], __shfl_xor(mx[hd], off, 32));
    if (lane == 0)
#pragma unroll
        for (int hd = 0; hd < NHEADS; ++hd) red[wave][hd] = mx[hd];
    __syncthreads();
    if (t < NHEADS) {
        float m = red[0][t];
#pragma unroll
        for (int w = 1; w < 8; ++w) m = fmaxf(m, red[w][t]);
        smax[t] = m;
    }
    __syncthreads();

    // ---- pass 2: per-head sum of exp ----
    float sm[NHEADS];
#pragma unroll
    for (int hd = 0; hd < NHEADS; ++hd) sm[hd] = 0.0f;
    for (int i = start + t; i < end; i += 256) {
        const float4* sp = (const float4*)(scores + (size_t)i * NHEADS);
        float4 s0 = sp[0], s1 = sp[1];
        sm[0] += expf(s0.x - smax[0]); sm[1] += expf(s0.y - smax[1]);
        sm[2] += expf(s0.z - smax[2]); sm[3] += expf(s0.w - smax[3]);
        sm[4] += expf(s1.x - smax[4]); sm[5] += expf(s1.y - smax[5]);
        sm[6] += expf(s1.z - smax[6]); sm[7] += expf(s1.w - smax[7]);
    }
#pragma unroll
    for (int hd = 0; hd < NHEADS; ++hd)
#pragma unroll
        for (int off = 16; off; off >>= 1)
            sm[hd] += __shfl_xor(sm[hd], off, 32);
    if (lane == 0)
#pragma unroll
        for (int hd = 0; hd < NHEADS; ++hd) red[wave][hd] = sm[hd];
    __syncthreads();
    if (t < NHEADS) {
        float s = 0.0f;
#pragma unroll
        for (int w = 0; w < 8; ++w) s += red[w][t];
        rden[t] = 1.0f / s;
    }
    __syncthreads();

    // ---- pass 3: head-averaged weights, weighted feature accumulation ----
    const int c = t & 63;              // float4 feature group: feats 4c..4c+3
    const int r = t >> 6;              // row phase within quad
    float ax = 0.0f, ay = 0.0f, az = 0.0f, aw = 0.0f;

    for (int base = start; base < end; base += 256) {
        const int i = base + t;
        float w = 0.0f;
        if (i < end) {
            const float4* sp = (const float4*)(scores + (size_t)i * NHEADS);
            float4 s0 = sp[0], s1 = sp[1];
            float s = expf(s0.x - smax[0]) * rden[0]
                    + expf(s0.y - smax[1]) * rden[1]
                    + expf(s0.z - smax[2]) * rden[2]
                    + expf(s0.w - smax[3]) * rden[3]
                    + expf(s1.x - smax[4]) * rden[4]
                    + expf(s1.y - smax[5]) * rden[5]
                    + expf(s1.z - smax[6]) * rden[6]
                    + expf(s1.w - smax[7]) * rden[7];
            w = s * (1.0f / NHEADS);
        }
        wbuf[t] = w;
        __syncthreads();
        const int cnt = min(256, end - base);
        for (int j = r; j < cnt; j += 4) {
            const float4 hv = *(const float4*)(h + (size_t)(base + j) * IN_FEAT + 4 * c);
            const float wj = wbuf[j];
            ax += hv.x * wj; ay += hv.y * wj; az += hv.z * wj; aw += hv.w * wj;
        }
        __syncthreads();
    }

    // combine the 4 row-phases per feature group
    accbuf[t * 4 + 0] = ax; accbuf[t * 4 + 1] = ay;
    accbuf[t * 4 + 2] = az; accbuf[t * 4 + 3] = aw;
    __syncthreads();
    if (r == 0) {
        float4 o;
        o.x = accbuf[c * 4 + 0] + accbuf[(64 + c) * 4 + 0] + accbuf[(128 + c) * 4 + 0] + accbuf[(192 + c) * 4 + 0];
        o.y = accbuf[c * 4 + 1] + accbuf[(64 + c) * 4 + 1] + accbuf[(128 + c) * 4 + 1] + accbuf[(192 + c) * 4 + 1];
        o.z = accbuf[c * 4 + 2] + accbuf[(64 + c) * 4 + 2] + accbuf[(128 + c) * 4 + 2] + accbuf[(192 + c) * 4 + 2];
        o.w = accbuf[c * 4 + 3] + accbuf[(64 + c) * 4 + 3] + accbuf[(128 + c) * 4 + 3] + accbuf[(192 + c) * 4 + 3];
        *(float4*)(out + (size_t)g * IN_FEAT + 4 * c) = o;
    }
}

// ---------------------------------------------------------------------------
extern "C" void kernel_launch(void* const* d_in, const int* in_sizes, int n_in,
                              void* d_out, int out_size, void* d_ws, size_t ws_size,
                              hipStream_t stream)
{
    const float* h   = (const float*)d_in[0];
    const int*   seg = (const int*)d_in[1];
    const float* W1  = (const float*)d_in[2];
    const float* b1  = (const float*)d_in[3];
    const float* W2  = (const float*)d_in[4];
    const float* b2  = (const float*)d_in[5];
    float*       out = (float*)d_out;

    const int N = in_sizes[1];             // number of nodes
    const int G = out_size / IN_FEAT;      // number of graphs

    // workspace layout: scores [N,8] f32 | W1p bf16 | W2p bf16
    char* ws = (char*)d_ws;
    float* scores = (float*)ws;
    size_t off = ((size_t)N * NHEADS * sizeof(float) + 255) & ~(size_t)255;
    __bf16* W1p = (__bf16*)(ws + off);
    off = (off + (size_t)8 * 8 * 32 * 16 * 2 + 255) & ~(size_t)255;
    __bf16* W2p = (__bf16*)(ws + off);

    const int packThreads = 8 * 8 * 32 * 16 + 4 * 32 * 16;
    pack_weights_kernel<<<(packThreads + 255) / 256, 256, 0, stream>>>(W1, W2, W1p, W2p);

    const int nBlocks = (N + 127) / 128;
    score_kernel<<<nBlocks, 256, 0, stream>>>(h, W1p, W2p, b1, b2, scores, N);

    pool_kernel<<<G, 256, 0, stream>>>(h, seg, scores, out, N);
}
